// SABlock_35055523070318
// MI455X (gfx1250) — matrix operand order repaired and checked by reading, not compiled, not run on hardware
//
#include <hip/hip_runtime.h>

// ---------------- problem constants ----------------
#define B_   4
#define S_   2048
#define D_   256
#define H_   4
#define DH_  64
#define ED_  1024            // EXPAND * D
#define NEG_ -1000000000.0f
#define EPS_ 0.001f

typedef __attribute__((ext_vector_type(16))) _Float16 v16h;
typedef __attribute__((ext_vector_type(8)))  float    v8f;

union Frag {                 // 8 VGPRs = 16 halves = 32 bytes
    v16h  h;
    uint4 u[2];
};

__device__ __forceinline__ float half_redmax(float v) {
#pragma unroll
    for (int m = 1; m < 16; m <<= 1) v = fmaxf(v, __shfl_xor(v, m, 32));
    return v;
}
__device__ __forceinline__ float half_redsum(float v) {
#pragma unroll
    for (int m = 1; m < 16; m <<= 1) v += __shfl_xor(v, m, 32);
    return v;
}

// ---- CDNA5 async copy: global -> LDS, tracked by ASYNCcnt ----
// ISA 15.18.3: copies memory into LDS at matching offsets; the
// instruction offset advances BOTH addresses, so two b128 ops move a 32B chunk.
__device__ __forceinline__ void async_copy_32B(void* lds_ptr, const void* gptr) {
    uint32_t lds = (uint32_t)(uintptr_t)lds_ptr;   // addr[31:0] == LDS offset
    asm volatile(
        "global_load_async_to_lds_b128 %0, %1, off\n\t"
        "global_load_async_to_lds_b128 %0, %1, off offset:16"
        :: "v"(lds), "v"(gptr) : "memory");
}
__device__ __forceinline__ void wait_async0() {
    asm volatile("s_wait_asynccnt 0x0" ::: "memory");
}

// ---------------- elementwise kernels ----------------
__global__ void bn_to_f16(const float* __restrict__ x,
                          const float* __restrict__ gamma,
                          const float* __restrict__ beta,
                          const float* __restrict__ mean,
                          const float* __restrict__ var,
                          _Float16* __restrict__ out, int total) {
    int i = blockIdx.x * blockDim.x + threadIdx.x;
    if (i >= total) return;
    int c = i & (D_ - 1);
    float v = (x[i] - mean[c]) * rsqrtf(var[c] + EPS_) * gamma[c] + beta[c];
    out[i] = (_Float16)v;
}

// W[K][N] f32  ->  Wt[N][K] f16 (contiguous over K)
__global__ void wconv_t(const float* __restrict__ W, _Float16* __restrict__ Wt,
                        int K, int N) {
    int i = blockIdx.x * blockDim.x + threadIdx.x;
    if (i >= K * N) return;
    int k = i / N, n = i - k * N;
    Wt[(size_t)n * K + k] = (_Float16)W[i];
}

// ---------------- WMMA GEMM, async double-buffered ----------------
#define BM 128
#define BN 128
#define BK 32
#define LDT 40               // padded LDS row stride (halves); 80B, 16B aligned

enum { EPI_QKV = 0, EPI_RES = 1, EPI_SWISH = 2 };

template <int EPI>
__global__ __launch_bounds__(256)
void gemm_wmma(const _Float16* __restrict__ A, const _Float16* __restrict__ Wt,
               int M, int N, int K,
               const float* __restrict__ res,    // residual (EPI_RES)
               float* __restrict__ outf,         // f32 out  (EPI_RES)
               _Float16* __restrict__ outh,      // f16 out  (EPI_SWISH)
               _Float16* __restrict__ qo, _Float16* __restrict__ ko,
               _Float16* __restrict__ vto)       // EPI_QKV targets
{
    __shared__ _Float16 sA[2][BM * LDT];
    __shared__ _Float16 sB[2][BN * LDT];

    const int tid  = threadIdx.x;
    const int lane = tid & 31;
    const int l    = lane & 15;
    const int sel  = lane >> 4;          // 0: lanes 0-15, 1: lanes 16-31
    const int wid  = tid >> 5;
    const int wm   = (wid & 3) * 32;     // wave M offset inside tile
    const int wn   = (wid >> 2) * 64;    // wave N offset inside tile
    const int m0   = blockIdx.x * BM;
    const int n0   = blockIdx.y * BN;

    v8f acc[2][4] = {};

    const int r = tid >> 1;              // 0..127
    const int c = (tid & 1) * 16;        // 0 or 16 (halves)

    const int nk = K / BK;
    // stage tile kk into LDS buffer `buf` via async DMA (32B per thread per array)
    auto issue = [&](int kk, int buf) {
        const int k0 = kk * BK;
        async_copy_32B(&sA[buf][r * LDT + c], A  + (size_t)(m0 + r) * K + k0 + c);
        async_copy_32B(&sB[buf][r * LDT + c], Wt + (size_t)(n0 + r) * K + k0 + c);
    };

    issue(0, 0);
    for (int kk = 0; kk < nk; ++kk) {
        const int buf = kk & 1;
        wait_async0();                   // this wave's tile-kk DMA done
        __syncthreads();                 // whole tile visible; prev compute done
        if (kk + 1 < nk) issue(kk + 1, buf ^ 1);   // overlap DMA with compute

        Frag a[2], b[4];
#pragma unroll
        for (int i = 0; i < 2; ++i) {    // A 16x32 layout
            const _Float16* p = &sA[buf][(wm + 16 * i + l) * LDT + sel * 8];
            a[i].u[0] = *(const uint4*)p;
            a[i].u[1] = *(const uint4*)(p + 16);
        }
#pragma unroll
        for (int j = 0; j < 4; ++j) {    // B 32x16 layout (from N-major Wt)
            const _Float16* p = &sB[buf][(wn + 16 * j + l) * LDT + sel * 8];
            b[j].u[0] = *(const uint4*)p;
            b[j].u[1] = *(const uint4*)(p + 16);
        }
#pragma unroll
        for (int i = 0; i < 2; ++i)
#pragma unroll
            for (int j = 0; j < 4; ++j)
                acc[i][j] = __builtin_amdgcn_wmma_f32_16x16x32_f16(
                    false, a[i].h, false, b[j].h, (short)0, acc[i][j], false, false);
    }

    // epilogue: C layout -> row = 16i + sel*8 + g, col = l
#pragma unroll
    for (int i = 0; i < 2; ++i)
#pragma unroll
        for (int j = 0; j < 4; ++j)
#pragma unroll
            for (int g = 0; g < 8; ++g) {
                const int m = m0 + wm + 16 * i + sel * 8 + g;
                const int n = n0 + wn + 16 * j + l;
                float v = acc[i][j][g];
                if constexpr (EPI == EPI_RES) {
                    outf[(size_t)m * N + n] = v + res[(size_t)m * N + n];
                } else if constexpr (EPI == EPI_SWISH) {
                    outh[(size_t)m * N + n] = (_Float16)(v / (1.0f + __expf(-v)));
                } else {                 // EPI_QKV scatter
                    const int bb = m >> 11;          // m / S_
                    const int s  = m & (S_ - 1);
                    const int hh = n / 192;          // 3*DH per head
                    const int rr = n - hh * 192;
                    const int part = rr >> 6;        // 0=q 1=k 2=v
                    const int d    = rr & 63;
                    const int bh   = bb * H_ + hh;
                    _Float16 hv = (_Float16)v;
                    if (part == 0)      qo [((size_t)bh * S_ + s) * DH_ + d] = hv;
                    else if (part == 1) ko [((size_t)bh * S_ + s) * DH_ + d] = hv;
                    else                vto[((size_t)bh * DH_ + d) * S_ + s] = hv;
                }
            }
}

// ---------------- flash attention (online softmax) ----------------
// grid: (S/128, B*H), block: 128 threads = 4 waves; each wave owns 32 q-rows.
__global__ __launch_bounds__(128)
void attn_flash(const _Float16* __restrict__ q, const _Float16* __restrict__ k,
                const _Float16* __restrict__ vt, const int* __restrict__ mask,
                _Float16* __restrict__ o16)
{
    __shared__ _Float16 sP[4][32 * LDT];   // wave-private P re-layout scratch

    const int tid  = threadIdx.x;
    const int lane = tid & 31;
    const int l    = lane & 15;
    const int sel  = lane >> 4;
    const int wid  = tid >> 5;
    const int bh   = blockIdx.y;
    const int bb   = bh >> 2;              // / H_
    const int hh   = bh & 3;
    const int q0   = blockIdx.x * 128 + wid * 32;
    const float scale = 0.0625f;           // D^-0.5 = 1/16

    const _Float16* Q  = q  + (size_t)bh * S_ * DH_;
    const _Float16* Kg = k  + (size_t)bh * S_ * DH_;
    const _Float16* Vt = vt + (size_t)bh * DH_ * S_;

    // Q fragments: 2 m-subtiles x 2 k-chunks (DH=64 -> two K=32 steps)
    Frag aQ[2][2];
#pragma unroll
    for (int i = 0; i < 2; ++i)
#pragma unroll
        for (int kk = 0; kk < 2; ++kk) {
            const _Float16* p = Q + (size_t)(q0 + 16 * i + l) * DH_ + 32 * kk + sel * 8;
            aQ[i][kk].u[0] = *(const uint4*)p;
            aQ[i][kk].u[1] = *(const uint4*)(p + 16);
        }

    v8f   o[2][4] = {};
    float rmax[2][8], rsum[2][8];
#pragma unroll
    for (int i = 0; i < 2; ++i)
#pragma unroll
        for (int g = 0; g < 8; ++g) { rmax[i][g] = -1e30f; rsum[i][g] = 0.0f; }

    for (int kb = 0; kb < S_; kb += 32) {
        // prefetch next K/V block while this one computes
        if (kb + 32 < S_) {
            __builtin_prefetch(Kg + (size_t)(kb + 32 + l) * DH_ + sel * 16, 0, 0);
            __builtin_prefetch(Vt + (size_t)(16 + l) * S_ + kb + 32 + sel * 16, 0, 0);
        }
        // ---- scores: S = Q * K^T for a 32x32 tile ----
        Frag bK[2][2];
#pragma unroll
        for (int j = 0; j < 2; ++j)
#pragma unroll
            for (int kk = 0; kk < 2; ++kk) {
                const _Float16* p = Kg + (size_t)(kb + 16 * j + l) * DH_ + 32 * kk + sel * 8;
                bK[j][kk].u[0] = *(const uint4*)p;
                bK[j][kk].u[1] = *(const uint4*)(p + 16);
            }
        float sv[2][2][8];
        float madd[2];
        madd[0] = mask[(size_t)bb * S_ + kb + l]      ? 0.0f : NEG_;
        madd[1] = mask[(size_t)bb * S_ + kb + 16 + l] ? 0.0f : NEG_;
#pragma unroll
        for (int i = 0; i < 2; ++i)
#pragma unroll
            for (int j = 0; j < 2; ++j) {
                v8f z = {};
                v8f sc = __builtin_amdgcn_wmma_f32_16x16x32_f16(
                    false, aQ[i][0].h, false, bK[j][0].h, (short)0, z, false, false);
                sc = __builtin_amdgcn_wmma_f32_16x16x32_f16(
                    false, aQ[i][1].h, false, bK[j][1].h, (short)0, sc, false, false);
#pragma unroll
                for (int g = 0; g < 8; ++g)
                    sv[i][j][g] = sc[g] * scale + madd[j];
            }

        // ---- online softmax update ----
#pragma unroll
        for (int i = 0; i < 2; ++i) {
            float corr[8];
#pragma unroll
            for (int g = 0; g < 8; ++g) {
                float mx = half_redmax(fmaxf(sv[i][0][g], sv[i][1][g]));
                float nm = fmaxf(rmax[i][g], mx);
                corr[g] = __expf(rmax[i][g] - nm);
                rmax[i][g] = nm;
                float p0 = __expf(sv[i][0][g] - nm);
                float p1 = __expf(sv[i][1][g] - nm);
                sv[i][0][g] = p0;
                sv[i][1][g] = p1;
                rsum[i][g] = rsum[i][g] * corr[g] + half_redsum(p0 + p1);
            }
#pragma unroll
            for (int jd = 0; jd < 4; ++jd)
#pragma unroll
                for (int g = 0; g < 8; ++g)
                    o[i][jd][g] *= corr[g];
            // C layout -> LDS (row-major P tile), wave-private so no barrier
#pragma unroll
            for (int j = 0; j < 2; ++j)
#pragma unroll
                for (int g = 0; g < 8; ++g)
                    sP[wid][(16 * i + sel * 8 + g) * LDT + 16 * j + l] =
                        (_Float16)sv[i][j][g];
        }

        // ---- O += P * V ----
        Frag aP[2];
#pragma unroll
        for (int i = 0; i < 2; ++i) {
            const _Float16* p = &sP[wid][(16 * i + l) * LDT + sel * 8];
            aP[i].u[0] = *(const uint4*)p;
            aP[i].u[1] = *(const uint4*)(p + 16);
        }
        Frag bV[4];
#pragma unroll
        for (int jd = 0; jd < 4; ++jd) {
            const _Float16* p = Vt + (size_t)(16 * jd + l) * S_ + kb + sel * 8;
            bV[jd].u[0] = *(const uint4*)p;
            bV[jd].u[1] = *(const uint4*)(p + 16);
        }
#pragma unroll
        for (int i = 0; i < 2; ++i)
#pragma unroll
            for (int jd = 0; jd < 4; ++jd)
                o[i][jd] = __builtin_amdgcn_wmma_f32_16x16x32_f16(
                    false, aP[i].h, false, bV[jd].h, (short)0, o[i][jd], false, false);
    }

    // ---- normalize and emit [B][S][D] f16 ----
#pragma unroll
    for (int i = 0; i < 2; ++i)
#pragma unroll
        for (int jd = 0; jd < 4; ++jd)
#pragma unroll
            for (int g = 0; g < 8; ++g) {
                const int s = q0 + 16 * i + sel * 8 + g;
                const int d = 16 * jd + l;
                float val = o[i][jd][g] / rsum[i][g];
                o16[((size_t)(bb * S_ + s)) * D_ + hh * DH_ + d] = (_Float16)val;
            }
}

// ---------------- host driver ----------------
extern "C" void kernel_launch(void* const* d_in, const int* in_sizes, int n_in,
                              void* d_out, int out_size, void* d_ws, size_t ws_size,
                              hipStream_t stream) {
    const float* x      = (const float*)d_in[0];
    const int*   mask   = (const int*)  d_in[1];
    const float* bn1g   = (const float*)d_in[2];
    const float* bn1b   = (const float*)d_in[3];
    const float* bn1m   = (const float*)d_in[4];
    const float* bn1v   = (const float*)d_in[5];
    const float* qkv_w  = (const float*)d_in[6];
    const float* proj_w = (const float*)d_in[7];
    const float* bn2g   = (const float*)d_in[8];
    const float* bn2b   = (const float*)d_in[9];
    const float* bn2m   = (const float*)d_in[10];
    const float* bn2v   = (const float*)d_in[11];
    const float* w1     = (const float*)d_in[12];
    const float* w2     = (const float*)d_in[13];
    float* out = (float*)d_out;

    const size_t MR = (size_t)B_ * S_;          // 8192 rows
    char* ws = (char*)d_ws;
    size_t off = 0;
    auto alloc = [&](size_t bytes) -> char* {
        char* p = ws + off;
        off = (off + bytes + 255) & ~(size_t)255;
        return p;
    };
    _Float16* h16    = (_Float16*)alloc(MR * D_  * 2);
    _Float16* qf     = (_Float16*)alloc(MR * D_  * 2);
    _Float16* kf     = (_Float16*)alloc(MR * D_  * 2);
    _Float16* vtf    = (_Float16*)alloc(MR * D_  * 2);
    _Float16* o16    = (_Float16*)alloc(MR * D_  * 2);
    float*    y1     = (float*)   alloc(MR * D_  * 4);
    _Float16* h2     = (_Float16*)alloc(MR * D_  * 2);
    _Float16* s16    = (_Float16*)alloc(MR * ED_ * 2);
    _Float16* qkvwT  = (_Float16*)alloc((size_t)3 * D_ * D_ * 2);
    _Float16* projwT = (_Float16*)alloc((size_t)D_ * D_ * 2);
    _Float16* w1T    = (_Float16*)alloc((size_t)D_ * ED_ * 2);
    _Float16* w2T    = (_Float16*)alloc((size_t)ED_ * D_ * 2);

    // weight conversion to transposed f16
    wconv_t<<<(D_ * 3 * D_ + 255) / 256, 256, 0, stream>>>(qkv_w, qkvwT, D_, 3 * D_);
    wconv_t<<<(D_ * D_      + 255) / 256, 256, 0, stream>>>(proj_w, projwT, D_, D_);
    wconv_t<<<(D_ * ED_     + 255) / 256, 256, 0, stream>>>(w1, w1T, D_, ED_);
    wconv_t<<<(ED_ * D_     + 255) / 256, 256, 0, stream>>>(w2, w2T, ED_, D_);

    const int total = (int)(MR * D_);
    // BN1 -> h16
    bn_to_f16<<<(total + 255) / 256, 256, 0, stream>>>(x, bn1g, bn1b, bn1m, bn1v, h16, total);

    // QKV GEMM (8192x768) with scatter epilogue
    {
        dim3 g(MR / BM, (3 * D_) / BN);
        gemm_wmma<EPI_QKV><<<g, 256, 0, stream>>>(h16, qkvwT, (int)MR, 3 * D_, D_,
                                                  nullptr, nullptr, nullptr, qf, kf, vtf);
    }
    // flash attention
    {
        dim3 g(S_ / 128, B_ * H_);
        attn_flash<<<g, 128, 0, stream>>>(qf, kf, vtf, mask, o16);
    }
    // proj GEMM + residual(x) -> y1 (f32)
    {
        dim3 g(MR / BM, D_ / BN);
        gemm_wmma<EPI_RES><<<g, 256, 0, stream>>>(o16, projwT, (int)MR, D_, D_,
                                                  x, y1, nullptr, nullptr, nullptr, nullptr);
    }
    // BN2 -> h2 (f16)
    bn_to_f16<<<(total + 255) / 256, 256, 0, stream>>>(y1, bn2g, bn2b, bn2m, bn2v, h2, total);
    // FFN1 + swish -> s16 (f16)
    {
        dim3 g(MR / BM, ED_ / BN);
        gemm_wmma<EPI_SWISH><<<g, 256, 0, stream>>>(h2, w1T, (int)MR, ED_, D_,
                                                    nullptr, nullptr, s16, nullptr, nullptr, nullptr);
    }
    // FFN2 + residual(y1) -> out (f32)
    {
        dim3 g(MR / BM, D_ / BN);
        gemm_wmma<EPI_RES><<<g, 256, 0, stream>>>(s16, w2T, (int)MR, D_, ED_,
                                                  y1, out, nullptr, nullptr, nullptr, nullptr);
    }
}
